// GATNet_38714835206277
// MI455X (gfx1250) — matrix operand order repaired and checked by reading, not compiled, not run on hardware
//
#include <hip/hip_runtime.h>
#include <math.h>

// ---------------------------------------------------------------------------
// Types for CDNA5 WMMA
// ---------------------------------------------------------------------------
typedef __attribute__((ext_vector_type(16))) __bf16 v16bf;
typedef __attribute__((ext_vector_type(8)))  float  v8f;

#define LRELU_ALPHA 0.2f

__device__ __forceinline__ float leaky(float x) { return x >= 0.f ? x : LRELU_ALPHA * x; }

// f32 -> bf16 round-to-nearest-even
__device__ __forceinline__ unsigned short f2bf(float f) {
    unsigned u = __float_as_uint(f);
    u += 0x7FFFu + ((u >> 16) & 1u);
    return (unsigned short)(u >> 16);
}

// order-preserving float<->uint encoding (for atomicMax on floats)
__device__ __forceinline__ unsigned encf(float f) {
    unsigned u = __float_as_uint(f);
    return (u & 0x80000000u) ? ~u : (u | 0x80000000u);
}
__device__ __forceinline__ float decf(unsigned u) {
    return __uint_as_float((u & 0x80000000u) ? (u & 0x7FFFFFFFu) : ~u);
}
#define ENC_NEG_INF 0x007FFFFFu  // encf(-inf)

// ---------------------------------------------------------------------------
// Small utility kernels
// ---------------------------------------------------------------------------
__global__ void fill_u32_kernel(unsigned* __restrict__ p, unsigned v, long n) {
    long i = (long)blockIdx.x * blockDim.x + threadIdx.x;
    if (i < n) p[i] = v;
}

__global__ void cvt_bf16_kernel(const float* __restrict__ in,
                                unsigned short* __restrict__ out, long n) {
    long i = (long)blockIdx.x * blockDim.x + threadIdx.x;
    if (i < n) out[i] = f2bf(in[i]);
}

// W1 [4][256][32] -> Wt1 [128][256] bf16, col c = h*32+f
__global__ void build_wt1_kernel(const float* __restrict__ W1,
                                 unsigned short* __restrict__ wt) {
    int idx = blockIdx.x * blockDim.x + threadIdx.x;
    if (idx >= 128 * 256) return;
    int c = idx >> 8, k = idx & 255;
    int h = c >> 5, f = c & 31;
    wt[idx] = f2bf(W1[((size_t)h * 256 + k) * 32 + f]);
}

// W2 [1][128][40] -> Wt2 [48][128] bf16 (cols 40..47 zero-padded)
__global__ void build_wt2_kernel(const float* __restrict__ W2,
                                 unsigned short* __restrict__ wt) {
    int idx = blockIdx.x * blockDim.x + threadIdx.x;
    if (idx >= 48 * 128) return;
    int c = idx >> 7, k = idx & 127;
    wt[idx] = (c < 40) ? f2bf(W2[(size_t)k * 40 + c]) : (unsigned short)0;
}

// h1 = leaky(agg) stored as bf16
__global__ void leaky_bf16_kernel(const float* __restrict__ in,
                                  unsigned short* __restrict__ out, long n) {
    long i = (long)blockIdx.x * blockDim.x + threadIdx.x;
    if (i < n) out[i] = f2bf(leaky(in[i]));
}

// ---------------------------------------------------------------------------
// bf16 WMMA GEMM: C[N, Ccols] = A[N, K] * Bt[Cpad, K]^T
// A row-major bf16, Bt column-of-B-major bf16 (Bt[c][k] = B[k][c]).
// One wave per 16-row tile; 16x16x32 WMMA over K; Cpad multiple of 16.
// ---------------------------------------------------------------------------
__global__ void gemm_bf16_wmma_kernel(const unsigned short* __restrict__ A,
                                      const unsigned short* __restrict__ Bt,
                                      float* __restrict__ C,
                                      int nRowTiles, int K, int Cpad,
                                      int Ccols, int ldc) {
    int warp = threadIdx.x >> 5;
    int lane = threadIdx.x & 31;
    int tile = blockIdx.x * (blockDim.x >> 5) + warp;
    if (tile >= nRowTiles) return;

    int  rowBase = tile << 4;
    int  mlane   = lane & 15;
    bool hi      = lane >= 16;

    // A-fragment base for this lane's row (ISA 16-bit A 16x32 layout)
    const unsigned short* Ar = A + (size_t)(rowBase + mlane) * K + (hi ? 8 : 0);

    union Frag { v16bf v; uint4 q[2]; };
    Frag af, bf;

    for (int c0 = 0; c0 < Cpad; c0 += 16) {
        v8f acc = {};
        const unsigned short* Br = Bt + (size_t)(c0 + mlane) * K + (hi ? 8 : 0);
        for (int k0 = 0; k0 < K; k0 += 32) {
            // A: lanes<16 get K {k0..k0+7, k0+16..k0+23}; lanes>=16 shifted by 8
            af.q[0] = *(const uint4*)(Ar + k0);
            af.q[1] = *(const uint4*)(Ar + k0 + 16);
            // B: lane n<16 -> col c0+n, K {k0..k0+7, k0+16..k0+23}; lanes>=16 shifted by 8
            bf.q[0] = *(const uint4*)(Br + k0);
            bf.q[1] = *(const uint4*)(Br + k0 + 16);
            acc = __builtin_amdgcn_wmma_f32_16x16x32_bf16(
                false, af.v, false, bf.v, (short)0, acc, false, false);
        }
        int col = c0 + mlane;
        if (col < Ccols) {
            int rb = rowBase + (hi ? 8 : 0);
#pragma unroll
            for (int v = 0; v < 8; ++v)
                C[(size_t)(rb + v) * ldc + col] = acc[v];
        }
    }
}

// ---------------------------------------------------------------------------
// Attention kernels (generic in H, F)
// ---------------------------------------------------------------------------
// e_dst[n,h] = dot(feat[n,h,:], a[h,:F]); e_src[n,h] = dot(feat[n,h,:], a[h,F:])
__global__ void scores_kernel(const float* __restrict__ feat,
                              const float* __restrict__ a,
                              float* __restrict__ edst, float* __restrict__ esrc,
                              int N, int H, int F) {
    int idx = blockIdx.x * blockDim.x + threadIdx.x;
    if (idx >= N * H) return;
    int n = idx / H, h = idx % H;
    const float* fp = feat + ((size_t)n * H + h) * F;
    const float* ad = a + (size_t)h * 2 * F;
    float sd = 0.f, ss = 0.f;
    for (int f = 0; f < F; ++f) {
        float v = fp[f];
        sd += v * ad[f];
        ss += v * ad[F + f];
    }
    edst[idx] = sd;
    esrc[idx] = ss;
}

// pass 1: e = leaky(edst[dst]+esrc[src]); store; segment-max via ordered uint
__global__ void edge_max_kernel(const int* __restrict__ src, const int* __restrict__ dst,
                                const float* __restrict__ edstv, const float* __restrict__ esrcv,
                                float* __restrict__ ew, unsigned* __restrict__ menc,
                                long EH, int H) {
    long idx = (long)blockIdx.x * blockDim.x + threadIdx.x;
    if (idx >= EH) return;
    long e = idx / H;
    int  h = (int)(idx % H);
    int  s = src[e], d = dst[e];
    float val = leaky(edstv[(size_t)d * H + h] + esrcv[(size_t)s * H + h]);
    ew[idx] = val;
    atomicMax(&menc[(size_t)d * H + h], encf(val));
}

// pass 2: w = exp(e - m[dst]); store; segment-sum into denom
__global__ void edge_exp_kernel(const int* __restrict__ dst,
                                float* __restrict__ ew,
                                const unsigned* __restrict__ menc,
                                float* __restrict__ denom, long EH, int H) {
    long idx = (long)blockIdx.x * blockDim.x + threadIdx.x;
    if (idx >= EH) return;
    long e = idx / H;
    int  h = (int)(idx % H);
    int  d = dst[e];
    float m = decf(menc[(size_t)d * H + h]);
    float w = __expf(ew[idx] - m);
    ew[idx] = w;
    atomicAdd(&denom[(size_t)d * H + h], w);
}

// pass 3: out[dst,h,f] += (w/denom[dst,h]) * feat[src,h,f]
__global__ void edge_agg_kernel(const int* __restrict__ src, const int* __restrict__ dst,
                                const float* __restrict__ ew, const float* __restrict__ denom,
                                const float* __restrict__ feat, float* __restrict__ out,
                                long total, int H, int F) {
    long idx = (long)blockIdx.x * blockDim.x + threadIdx.x;
    if (idx >= total) return;
    int  HF = H * F;
    long e  = idx / HF;
    int  r  = (int)(idx % HF);
    int  h  = r / F;
    int  s = src[e], d = dst[e];
    float attn = ew[(size_t)e * H + h] / fmaxf(denom[(size_t)d * H + h], 1e-16f);
    atomicAdd(&out[(size_t)d * HF + r], attn * feat[(size_t)s * HF + r]);
}

__global__ void logsoftmax_kernel(float* __restrict__ out, int N, int C) {
    int n = blockIdx.x * blockDim.x + threadIdx.x;
    if (n >= N) return;
    float* p = out + (size_t)n * C;
    float m = -INFINITY;
    for (int c = 0; c < C; ++c) m = fmaxf(m, p[c]);
    float s = 0.f;
    for (int c = 0; c < C; ++c) s += __expf(p[c] - m);
    float l = m + logf(s);
    for (int c = 0; c < C; ++c) p[c] -= l;
}

// ---------------------------------------------------------------------------
// Host-side launch
// ---------------------------------------------------------------------------
static inline unsigned nblk(long n, int t) { return (unsigned)((n + t - 1) / t); }

extern "C" void kernel_launch(void* const* d_in, const int* in_sizes, int n_in,
                              void* d_out, int out_size, void* d_ws, size_t ws_size,
                              hipStream_t stream) {
    const int N = 50000, DIN = 256, H1 = 4, F1 = 32, HF1 = 128;
    const int C2 = 40, CPAD2 = 48, K2 = 128;
    const int E = in_sizes[1] / 2;

    const float* x  = (const float*)d_in[0];
    const int*   ei = (const int*)d_in[1];
    const float* W1 = (const float*)d_in[2];
    const float* a1 = (const float*)d_in[3];
    const float* W2 = (const float*)d_in[4];
    const float* a2 = (const float*)d_in[5];
    const int* src = ei;
    const int* dst = ei + E;
    float* outp = (float*)d_out;

    // workspace layout
    char* ws = (char*)d_ws;
    size_t off = 0;
    auto carve = [&](size_t bytes) {
        size_t r = off;
        off += (bytes + 255) & ~(size_t)255;
        return r;
    };
    unsigned short* xbf    = (unsigned short*)(ws + carve((size_t)N * DIN * 2)); // reused as h1bf
    unsigned short* wt1    = (unsigned short*)(ws + carve(128 * 256 * 2));
    unsigned short* wt2    = (unsigned short*)(ws + carve(48 * 128 * 2));
    float*          hfeat1 = (float*)(ws + carve((size_t)N * HF1 * 4));
    float*          edst_b = (float*)(ws + carve((size_t)N * H1 * 4));
    float*          esrc_b = (float*)(ws + carve((size_t)N * H1 * 4));
    float*          ew     = (float*)(ws + carve((size_t)E * H1 * 4));
    unsigned*       menc   = (unsigned*)(ws + carve((size_t)N * H1 * 4));
    float*          denom  = (float*)(ws + carve((size_t)N * H1 * 4));
    float*          agg1   = (float*)(ws + carve((size_t)N * HF1 * 4));
    float*          hfeat2 = (float*)(ws + carve((size_t)N * C2 * 4));
    unsigned short* h1bf   = xbf;  // reuse: xbf dead after gemm1
    (void)ws_size; (void)n_in; (void)out_size;

    const int T = 256;
    const int rowTiles = N / 16;                    // 3125, exact
    const unsigned gemmBlocks = nblk(rowTiles, 8);  // 8 waves per block

    // ---- prep -------------------------------------------------------------
    cvt_bf16_kernel<<<nblk((long)N * DIN, T), T, 0, stream>>>(x, xbf, (long)N * DIN);
    build_wt1_kernel<<<nblk(128 * 256, T), T, 0, stream>>>(W1, wt1);
    build_wt2_kernel<<<nblk(48 * 128, T), T, 0, stream>>>(W2, wt2);

    // ---- layer 1 ----------------------------------------------------------
    gemm_bf16_wmma_kernel<<<gemmBlocks, T, 0, stream>>>(xbf, wt1, hfeat1,
                                                        rowTiles, DIN, HF1, HF1, HF1);
    scores_kernel<<<nblk((long)N * H1, T), T, 0, stream>>>(hfeat1, a1, edst_b, esrc_b, N, H1, F1);
    fill_u32_kernel<<<nblk((long)N * H1, T), T, 0, stream>>>(menc, ENC_NEG_INF, (long)N * H1);
    fill_u32_kernel<<<nblk((long)N * H1, T), T, 0, stream>>>((unsigned*)denom, 0u, (long)N * H1);
    fill_u32_kernel<<<nblk((long)N * HF1, T), T, 0, stream>>>((unsigned*)agg1, 0u, (long)N * HF1);
    edge_max_kernel<<<nblk((long)E * H1, T), T, 0, stream>>>(src, dst, edst_b, esrc_b, ew, menc,
                                                             (long)E * H1, H1);
    edge_exp_kernel<<<nblk((long)E * H1, T), T, 0, stream>>>(dst, ew, menc, denom,
                                                             (long)E * H1, H1);
    edge_agg_kernel<<<nblk((long)E * HF1, T), T, 0, stream>>>(src, dst, ew, denom, hfeat1, agg1,
                                                              (long)E * HF1, H1, F1);
    leaky_bf16_kernel<<<nblk((long)N * HF1, T), T, 0, stream>>>(agg1, h1bf, (long)N * HF1);

    // ---- layer 2 ----------------------------------------------------------
    gemm_bf16_wmma_kernel<<<gemmBlocks, T, 0, stream>>>(h1bf, wt2, hfeat2,
                                                        rowTiles, K2, CPAD2, C2, C2);
    scores_kernel<<<nblk((long)N, T), T, 0, stream>>>(hfeat2, a2, edst_b, esrc_b, N, 1, C2);
    fill_u32_kernel<<<nblk((long)N, T), T, 0, stream>>>(menc, ENC_NEG_INF, (long)N);
    fill_u32_kernel<<<nblk((long)N, T), T, 0, stream>>>((unsigned*)denom, 0u, (long)N);
    fill_u32_kernel<<<nblk((long)N * C2, T), T, 0, stream>>>((unsigned*)outp, 0u, (long)N * C2);
    edge_max_kernel<<<nblk((long)E, T), T, 0, stream>>>(src, dst, edst_b, esrc_b, ew, menc,
                                                        (long)E, 1);
    edge_exp_kernel<<<nblk((long)E, T), T, 0, stream>>>(dst, ew, menc, denom, (long)E, 1);
    edge_agg_kernel<<<nblk((long)E * C2, T), T, 0, stream>>>(src, dst, ew, denom, hfeat2, outp,
                                                             (long)E * C2, 1, C2);
    logsoftmax_kernel<<<nblk((long)N, T), T, 0, stream>>>(outp, N, C2);
}
